// SCGEncoder_56478819942799
// MI455X (gfx1250) — compile-verified, matrix operand order checked by reading, not run on hardware
//
#include <hip/hip_runtime.h>
#include <hip/hip_bf16.h>

typedef _Float16 half_t;
typedef _Float16 v16h __attribute__((ext_vector_type(16)));
typedef _Float16 v8h  __attribute__((ext_vector_type(8)));
typedef float    v8f  __attribute__((ext_vector_type(8)));
typedef unsigned int u32x4 __attribute__((ext_vector_type(4)));

#define BN_EPS 1e-5f
#define NCHUNK 256

// ---------------------------------------------------------------------------
// Farthest point sampling: one block per batch element. dist[] in LDS,
// argmax with lowest-index tie-break to match jnp.argmax semantics.
// pts element (b, j, c) at  pts + b*bstride + j*pstride + c*cstride
// ---------------------------------------------------------------------------
__global__ void fps_kernel(const float* __restrict__ pts, int bstride, int pstride,
                           int cstride, int n, int npoint,
                           int* __restrict__ fps_idx, float* __restrict__ new_xyz) {
  const int b   = blockIdx.x;
  const int tid = threadIdx.x;
  const float* base = pts + (size_t)b * bstride;
  __shared__ float dist[2048];
  __shared__ float sval[256];
  __shared__ int   sidx[256];
  for (int j = tid; j < n; j += blockDim.x) dist[j] = 1e10f;
  __syncthreads();
  int far = 0;
  for (int it = 0; it < npoint; ++it) {
    float cx = base[(size_t)far * pstride + 0 * cstride];
    float cy = base[(size_t)far * pstride + 1 * cstride];
    float cz = base[(size_t)far * pstride + 2 * cstride];
    if (tid == 0) {
      fps_idx[b * npoint + it] = far;
      float* nx = new_xyz + ((size_t)b * npoint + it) * 3;
      nx[0] = cx; nx[1] = cy; nx[2] = cz;
    }
    float bv = -1.0f; int bi = 0x7fffffff;
    for (int j = tid; j < n; j += blockDim.x) {
      float dx = base[(size_t)j * pstride + 0 * cstride] - cx;
      float dy = base[(size_t)j * pstride + 1 * cstride] - cy;
      float dz = base[(size_t)j * pstride + 2 * cstride] - cz;
      float d  = dx * dx + dy * dy + dz * dz;
      float nd = fminf(dist[j], d);
      dist[j] = nd;
      if (nd > bv || (nd == bv && j < bi)) { bv = nd; bi = j; }
    }
    sval[tid] = bv; sidx[tid] = bi;
    __syncthreads();
    for (int s = blockDim.x / 2; s > 0; s >>= 1) {
      if (tid < s) {
        if (sval[tid + s] > sval[tid] ||
            (sval[tid + s] == sval[tid] && sidx[tid + s] < sidx[tid])) {
          sval[tid] = sval[tid + s]; sidx[tid] = sidx[tid + s];
        }
      }
      __syncthreads();
    }
    far = sidx[0];
    __syncthreads();
  }
}

// ---------------------------------------------------------------------------
// Ball query: first K in-radius indices (ascending), pad with first hit.
// ---------------------------------------------------------------------------
__global__ void ball_query_kernel(const float* __restrict__ pts, int bstride, int pstride,
                                  int cstride, const float* __restrict__ new_xyz,
                                  int n, int S, int K, float r2,
                                  int* __restrict__ out, int total) {
  int t = blockIdx.x * blockDim.x + threadIdx.x;
  if (t >= total) return;
  int b = t / S, s = t % S;
  const float* q = new_xyz + ((size_t)b * S + s) * 3;
  float qx = q[0], qy = q[1], qz = q[2];
  const float* base = pts + (size_t)b * bstride;
  int* o = out + ((size_t)b * S + s) * K;
  int cnt = 0, first = -1;
  for (int j = 0; j < n && cnt < K; ++j) {
    float dx = base[(size_t)j * pstride + 0 * cstride] - qx;
    float dy = base[(size_t)j * pstride + 1 * cstride] - qy;
    float dz = base[(size_t)j * pstride + 2 * cstride] - qz;
    float d2 = dx * dx + dy * dy + dz * dz;
    if (d2 <= r2) { if (first < 0) first = j; o[cnt++] = j; }
  }
  if (first < 0) first = 0;
  for (; cnt < K; ++cnt) o[cnt] = first;
}

// ---------------------------------------------------------------------------
// Grouping -> point-major FP16 X[M][Cpad], zero-padded channels.
// m = (b*S+s)*K + k
// ---------------------------------------------------------------------------
__global__ void group1_kernel(const float* __restrict__ in0, const float* __restrict__ new_xyz,
                              const int* __restrict__ ballidx, half_t* __restrict__ X,
                              int B, int N, int S, int K) {
  size_t M = (size_t)B * S * K;
  size_t m = (size_t)blockIdx.x * blockDim.x + threadIdx.x;
  if (m >= M) return;
  size_t p = m / K; int s = (int)(p % S); int b = (int)(p / S);
  int j = ballidx[m];
  const float* q = new_xyz + ((size_t)b * S + s) * 3;
  const float* base = in0 + (size_t)b * 6 * N;
  half_t* x = X + m * 32;                 // Cpad = 32
  x[0] = (half_t)(base[0 * (size_t)N + j] - q[0]);
  x[1] = (half_t)(base[1 * (size_t)N + j] - q[1]);
  x[2] = (half_t)(base[2 * (size_t)N + j] - q[2]);
  x[3] = (half_t)base[3 * (size_t)N + j];
  x[4] = (half_t)base[4 * (size_t)N + j];
  x[5] = (half_t)base[5 * (size_t)N + j];
  for (int c = 6; c < 32; ++c) x[c] = (half_t)0.0f;
}

__global__ void group2_kernel(const float* __restrict__ xyz_pts /*[b][n][3]*/,
                              const half_t* __restrict__ feat /*[B*n][128]*/,
                              const float* __restrict__ new_xyz /*[b][S][3]*/,
                              const int* __restrict__ ballidx, half_t* __restrict__ X,
                              int B, int n, int S, int K) {
  size_t M = (size_t)B * S * K;
  size_t m = (size_t)blockIdx.x * blockDim.x + threadIdx.x;
  if (m >= M) return;
  size_t p = m / K; int s = (int)(p % S); int b = (int)(p / S);
  int j = ballidx[m];
  const float* q  = new_xyz + ((size_t)b * S + s) * 3;
  const float* pt = xyz_pts + ((size_t)b * n + j) * 3;
  half_t* x = X + m * 160;                // Cpad = 160 (3 + 128 + 29 pad)
  x[0] = (half_t)(pt[0] - q[0]);
  x[1] = (half_t)(pt[1] - q[1]);
  x[2] = (half_t)(pt[2] - q[2]);
  const half_t* f = feat + ((size_t)b * n + j) * 128;
  for (int c = 0; c < 128; ++c) x[3 + c] = f[c];
  for (int c = 131; c < 160; ++c) x[c] = (half_t)0.0f;
}

__global__ void group3_kernel(const float* __restrict__ xyz2 /*[b][n][3]*/,
                              const half_t* __restrict__ feat /*[B*n][256]*/,
                              half_t* __restrict__ X, int B, int n) {
  size_t M = (size_t)B * n;
  size_t m = (size_t)blockIdx.x * blockDim.x + threadIdx.x;
  if (m >= M) return;
  half_t* x = X + m * 288;                // Cpad = 288 (3 + 256 + 29 pad)
  x[0] = (half_t)xyz2[m * 3 + 0];
  x[1] = (half_t)xyz2[m * 3 + 1];
  x[2] = (half_t)xyz2[m * 3 + 2];
  const half_t* f = feat + m * 256;
  for (int c = 0; c < 256; ++c) x[3 + c] = f[c];
  for (int c = 259; c < 288; ++c) x[c] = (half_t)0.0f;
}

// Convert f32 weights (Cout x Cin) to zero-padded f16 (Cout x Kpad)
__global__ void convert_w_kernel(const float* __restrict__ W, half_t* __restrict__ Wh,
                                 int Cout, int Cin, int Kpad) {
  int total = Cout * Kpad;
  int i = blockIdx.x * blockDim.x + threadIdx.x;
  if (i >= total) return;
  int o = i / Kpad, k = i % Kpad;
  Wh[i] = (k < Cin) ? (half_t)W[(size_t)o * Cin + k] : (half_t)0.0f;
}

// ---------------------------------------------------------------------------
// WMMA GEMM: Y[M][Cout] = X[M][Kpad] * Wh(Cout x Kpad)^T + bias, F32 acc.
// One wave per block; 16 rows x 128 cols per wave (8 accumulators share one A
// fragment). Fragment element pair mapping (ISA 7.12.2, 16-bit A/B 16x32):
//   VGPR v holds K = (v>=4?16:0) + (v&3)*2 + hf*8 + {0,1}
// which is contiguous within each q-half -> two b128 loads per fragment.
// ---------------------------------------------------------------------------
__global__ void wmma_gemm_kernel(const half_t* __restrict__ Wh, const float* __restrict__ bias,
                                 const half_t* __restrict__ X, half_t* __restrict__ Y,
                                 int Cout, int Kpad, int M) {
  const int lane = threadIdx.x;      // 0..31 (wave32)
  const int nrow = lane & 15;
  const int hf   = lane >> 4;
  const int r0   = blockIdx.y * 16;
  const int c0   = blockIdx.x * 128;
  union Frag { v16h h; u32x4 q[2]; };
  v8f acc[8] = {v8f{}, v8f{}, v8f{}, v8f{}, v8f{}, v8f{}, v8f{}, v8f{}};
  const half_t* wrow  = Wh + (size_t)(r0 + nrow) * Kpad + hf * 8;
  const half_t* xbase = X + (size_t)(c0 + nrow) * Kpad + hf * 8;
  for (int k0 = 0; k0 < Kpad; k0 += 32) {
    Frag a;
    a.q[0] = *(const u32x4*)(wrow + k0);
    a.q[1] = *(const u32x4*)(wrow + k0 + 16);
#pragma unroll
    for (int t = 0; t < 8; ++t) {
      Frag bm;
      const half_t* xc = xbase + (size_t)t * 16 * Kpad;
      bm.q[0] = *(const u32x4*)(xc + k0);
      bm.q[1] = *(const u32x4*)(xc + k0 + 16);
      acc[t] = __builtin_amdgcn_wmma_f32_16x16x32_f16(
          false, a.h, false, bm.h, (short)0, acc[t], false, false);
    }
  }
#pragma unroll
  for (int t = 0; t < 8; ++t) {
    v8h o;
#pragma unroll
    for (int v = 0; v < 8; ++v) {
      int row = r0 + v + 8 * hf;              // D layout: VGPR v, half hf
      o[v] = (half_t)(acc[t][v] + bias[row]);
    }
    *(v8h*)(Y + (size_t)(c0 + t * 16 + nrow) * Cout + r0 + 8 * hf) = o;
  }
}

// ---------------------------------------------------------------------------
// Deterministic BN statistics in two stages over NCHUNK fixed m-chunks.
// Y is [M][C].
// ---------------------------------------------------------------------------
__global__ void bn_stats1_kernel(const half_t* __restrict__ Y, float* __restrict__ psum,
                                 float* __restrict__ psqs, int C, int M) {
  const int chunk = blockIdx.x;
  const int mlen  = M / NCHUNK;
  const int m0    = chunk * mlen;
  const int t     = threadIdx.x;
  if (C >= 256) {
    int nc = C / 256;                    // 1, 2 or 4
    float s[4] = {0.f, 0.f, 0.f, 0.f}, q[4] = {0.f, 0.f, 0.f, 0.f};
    for (int m = m0; m < m0 + mlen; ++m) {
      const half_t* row = Y + (size_t)m * C;
      for (int i = 0; i < nc; ++i) {
        float v = (float)row[t + i * 256];
        s[i] += v; q[i] += v * v;
      }
    }
    for (int i = 0; i < nc; ++i) {
      psum[(size_t)chunk * C + t + i * 256] = s[i];
      psqs[(size_t)chunk * C + t + i * 256] = q[i];
    }
  } else {
    int G = 256 / C;                     // C = 64 -> 4, C = 128 -> 2
    int g = t / C, c = t % C;
    int sub = mlen / G;
    float s = 0.f, q = 0.f;
    for (int m = m0 + g * sub; m < m0 + (g + 1) * sub; ++m) {
      float v = (float)Y[(size_t)m * C + c];
      s += v; q += v * v;
    }
    __shared__ float sh1[256], sh2[256];
    sh1[t] = s; sh2[t] = q;
    __syncthreads();
    if (g == 0) {
      for (int gg = 1; gg < G; ++gg) { s += sh1[gg * C + c]; q += sh2[gg * C + c]; }
      psum[(size_t)chunk * C + c] = s;
      psqs[(size_t)chunk * C + c] = q;
    }
  }
}

__global__ void bn_stats2_kernel(const float* __restrict__ psum, const float* __restrict__ psqs,
                                 float* __restrict__ mean, float* __restrict__ var,
                                 int C, int M) {
  int c = blockIdx.x * blockDim.x + threadIdx.x;
  if (c >= C) return;
  float s = 0.f, q = 0.f;
  for (int ch = 0; ch < NCHUNK; ++ch) {
    s += psum[(size_t)ch * C + c];
    q += psqs[(size_t)ch * C + c];
  }
  float mu = s / (float)M;
  mean[c] = mu;
  var[c]  = q / (float)M - mu * mu;
}

__global__ void bn_relu_kernel(half_t* __restrict__ Y, const float* __restrict__ mean,
                               const float* __restrict__ var, const float* __restrict__ g,
                               const float* __restrict__ beta, int C, size_t total) {
  size_t i = (size_t)blockIdx.x * blockDim.x + threadIdx.x;
  if (i >= total) return;
  int c = (int)(i % C);
  float v = (float)Y[i];
  float r = g[c] * (v - mean[c]) * rsqrtf(var[c] + BN_EPS) + beta[c];
  Y[i] = (half_t)fmaxf(r, 0.0f);
}

// Max over nsample axis: Y[(p*K+k)][C] -> out[p][C]
__global__ void maxpool_kernel(const half_t* __restrict__ Y, half_t* __restrict__ out,
                               int C, int P, int K) {
  size_t total = (size_t)P * C;
  size_t i = (size_t)blockIdx.x * blockDim.x + threadIdx.x;
  if (i >= total) return;
  int p = (int)(i / C), c = (int)(i % C);
  const half_t* r = Y + (size_t)p * K * C + c;
  float mx = -1e30f;
  for (int k = 0; k < K; ++k) mx = fmaxf(mx, (float)r[(size_t)k * C]);
  out[i] = (half_t)mx;
}

// inv[m] = 1 / max(||H[m][:]||2, 1e-12), H is [M][P]
__global__ void colnorm_kernel(const half_t* __restrict__ H, float* __restrict__ inv,
                               int P, int M) {
  int m = blockIdx.x * blockDim.x + threadIdx.x;
  if (m >= M) return;
  const half_t* r = H + (size_t)m * P;
  float s = 0.f;
  for (int p = 0; p < P; ++p) { float v = (float)r[p]; s += v * v; }
  inv[m] = 1.0f / fmaxf(sqrtf(s), 1e-12f);
}

// out[b, off+p] = max_s H[(b*S+s)][p] * inv[b*S+s]
__global__ void rowmax_kernel(const half_t* __restrict__ H, const float* __restrict__ inv,
                              float* __restrict__ out, int B, int P, int S,
                              int off, int ostride) {
  int t = blockIdx.x * blockDim.x + threadIdx.x;
  if (t >= B * P) return;
  int b = t / P, p = t % P;
  const half_t* r = H + (size_t)b * S * P + p;
  const float* iv = inv + (size_t)b * S;
  float mx = -1e30f;
  for (int s = 0; s < S; ++s) mx = fmaxf(mx, (float)r[(size_t)s * P] * iv[s]);
  out[(size_t)b * ostride + off + p] = mx;
}

// ---------------------------------------------------------------------------
// Host-side drivers
// ---------------------------------------------------------------------------
static inline int pad32(int c) { return (c + 31) & ~31; }

static void run_gemm(const float* W, const float* bias, half_t* wbuf,
                     const half_t* X, half_t* Y,
                     int Cout, int Cin, int M, hipStream_t s) {
  int Kpad = pad32(Cin);
  int wtotal = Cout * Kpad;
  convert_w_kernel<<<(wtotal + 255) / 256, 256, 0, s>>>(W, wbuf, Cout, Cin, Kpad);
  dim3 grid(M / 128, Cout / 16);
  wmma_gemm_kernel<<<grid, dim3(32), 0, s>>>(wbuf, bias, X, Y, Cout, Kpad, M);
}

static void run_bn_relu(half_t* Y, const float* g, const float* beta,
                        float* mean, float* var, float* psum, float* psqs,
                        int C, int M, hipStream_t s) {
  bn_stats1_kernel<<<NCHUNK, 256, 0, s>>>(Y, psum, psqs, C, M);
  bn_stats2_kernel<<<(C + 255) / 256, 256, 0, s>>>(psum, psqs, mean, var, C, M);
  size_t total = (size_t)C * M;
  bn_relu_kernel<<<(unsigned)((total + 255) / 256), 256, 0, s>>>(Y, mean, var, g, beta, C, total);
}

extern "C" void kernel_launch(void* const* d_in, const int* in_sizes, int n_in,
                              void* d_out, int out_size, void* d_ws, size_t ws_size,
                              hipStream_t stream) {
  (void)out_size; (void)ws_size;
  const float* in0 = (const float*)d_in[0];
  const float* P[64];
  for (int i = 0; i < n_in && i < 64; ++i) P[i] = (const float*)d_in[i];

  // Resolve param flattening order: dict insertion (sa1 first, w=(64,6)->384)
  // vs jax-sorted (pm1 first, w1=(512,128)->65536).
  int i_sa1, i_sa2, i_sa3, i_pm1, i_pm2, i_pm3;
  if (in_sizes[1] == 384) { i_sa1 = 1; i_sa2 = 13; i_sa3 = 25; i_pm1 = 37; i_pm2 = 43; i_pm3 = 49; }
  else                    { i_pm1 = 1; i_pm2 = 7;  i_pm3 = 13; i_sa1 = 19; i_sa2 = 31; i_sa3 = 43; }

  const int B = 32, N = 2048;
  const int S1 = 512, K1 = 48;
  const int S2 = 128, K2 = 64;
  const int M1 = B * S1 * K1;      // 786432
  const int M2 = B * S2 * K2;      // 262144
  const int Mp1 = B * S1;          // 16384
  const int Mp2 = B * S2;          // 4096
  const int M3  = B * S2;          // 4096

  // Workspace carve-up
  char* ws = (char*)d_ws;
  size_t off = 0;
  auto take = [&](size_t bytes) -> void* {
    void* p = ws + off;
    off = (off + bytes + 255) & ~(size_t)255;
    return p;
  };
  int*    fps1   = (int*)take((size_t)B * S1 * 4);
  float*  xyz1   = (float*)take((size_t)B * S1 * 3 * 4);
  int*    ball1  = (int*)take((size_t)B * S1 * K1 * 4);
  int*    fps2   = (int*)take((size_t)B * S2 * 4);
  float*  xyz2   = (float*)take((size_t)B * S2 * 3 * 4);
  int*    ball2  = (int*)take((size_t)B * S2 * K2 * 4);
  half_t* l1feat = (half_t*)take((size_t)Mp1 * 128 * 2);
  half_t* l2feat = (half_t*)take((size_t)Mp2 * 256 * 2);
  float*  mean   = (float*)take(1024 * 4);
  float*  var    = (float*)take(1024 * 4);
  float*  inv    = (float*)take((size_t)Mp1 * 4);
  float*  psum   = (float*)take((size_t)NCHUNK * 1024 * 4);
  float*  psqs   = (float*)take((size_t)NCHUNK * 1024 * 4);
  half_t* wbuf   = (half_t*)take((size_t)1024 * 1024 * 2);
  half_t* bufA   = (half_t*)take((size_t)M1 * 128 * 2);
  half_t* bufB   = (half_t*)take((size_t)M1 * 128 * 2);
  float*  out    = (float*)d_out;

  // ---------------- SA1 ----------------
  fps_kernel<<<B, 256, 0, stream>>>(in0, 6 * N, 1, N, N, S1, fps1, xyz1);
  ball_query_kernel<<<(B * S1 + 255) / 256, 256, 0, stream>>>(
      in0, 6 * N, 1, N, xyz1, N, S1, K1, 0.23f * 0.23f, ball1, B * S1);
  group1_kernel<<<(M1 + 255) / 256, 256, 0, stream>>>(in0, xyz1, ball1, bufA, B, N, S1, K1);
  {
    int cin = 6; const int couts[3] = {64, 64, 128};
    half_t* cur = bufA; half_t* nxt = bufB;
    for (int l = 0; l < 3; ++l) {
      run_gemm(P[i_sa1 + 4 * l], P[i_sa1 + 4 * l + 1], wbuf, cur, nxt, couts[l], cin, M1, stream);
      run_bn_relu(nxt, P[i_sa1 + 4 * l + 2], P[i_sa1 + 4 * l + 3], mean, var, psum, psqs,
                  couts[l], M1, stream);
      half_t* t = cur; cur = nxt; nxt = t; cin = couts[l];
    }
    maxpool_kernel<<<(unsigned)(((size_t)Mp1 * 128 + 255) / 256), 256, 0, stream>>>(
        cur, l1feat, 128, Mp1, K1);
  }
  // ---------------- pred1 ----------------
  run_gemm(P[i_pm1 + 0], P[i_pm1 + 1], wbuf, l1feat, bufA, 512, 128, Mp1, stream);
  run_bn_relu(bufA, P[i_pm1 + 2], P[i_pm1 + 3], mean, var, psum, psqs, 512, Mp1, stream);
  run_gemm(P[i_pm1 + 4], P[i_pm1 + 5], wbuf, bufA, bufB, 512, 512, Mp1, stream);
  colnorm_kernel<<<(Mp1 + 255) / 256, 256, 0, stream>>>(bufB, inv, 512, Mp1);
  rowmax_kernel<<<(B * 512 + 255) / 256, 256, 0, stream>>>(bufB, inv, out, B, 512, S1, 0, 1536);

  // ---------------- SA2 ----------------
  fps_kernel<<<B, 256, 0, stream>>>(xyz1, S1 * 3, 3, 1, S1, S2, fps2, xyz2);
  ball_query_kernel<<<(B * S2 + 255) / 256, 256, 0, stream>>>(
      xyz1, S1 * 3, 3, 1, xyz2, S1, S2, K2, 0.4f * 0.4f, ball2, B * S2);
  group2_kernel<<<(M2 + 255) / 256, 256, 0, stream>>>(
      xyz1, l1feat, xyz2, ball2, bufA, B, S1, S2, K2);
  {
    int cin = 131; const int couts[3] = {128, 128, 256};
    half_t* cur = bufA; half_t* nxt = bufB;
    for (int l = 0; l < 3; ++l) {
      run_gemm(P[i_sa2 + 4 * l], P[i_sa2 + 4 * l + 1], wbuf, cur, nxt, couts[l], cin, M2, stream);
      run_bn_relu(nxt, P[i_sa2 + 4 * l + 2], P[i_sa2 + 4 * l + 3], mean, var, psum, psqs,
                  couts[l], M2, stream);
      half_t* t = cur; cur = nxt; nxt = t; cin = couts[l];
    }
    maxpool_kernel<<<(unsigned)(((size_t)Mp2 * 256 + 255) / 256), 256, 0, stream>>>(
        cur, l2feat, 256, Mp2, K2);
  }
  // ---------------- pred2 ----------------
  run_gemm(P[i_pm2 + 0], P[i_pm2 + 1], wbuf, l2feat, bufA, 512, 256, Mp2, stream);
  run_bn_relu(bufA, P[i_pm2 + 2], P[i_pm2 + 3], mean, var, psum, psqs, 512, Mp2, stream);
  run_gemm(P[i_pm2 + 4], P[i_pm2 + 5], wbuf, bufA, bufB, 512, 512, Mp2, stream);
  colnorm_kernel<<<(Mp2 + 255) / 256, 256, 0, stream>>>(bufB, inv, 512, Mp2);
  rowmax_kernel<<<(B * 512 + 255) / 256, 256, 0, stream>>>(bufB, inv, out, B, 512, S2, 512, 1536);

  // ---------------- SA3 (group_all) ----------------
  group3_kernel<<<(M3 + 255) / 256, 256, 0, stream>>>(xyz2, l2feat, bufA, B, S2);
  {
    int cin = 259; const int couts[3] = {256, 512, 1024};
    half_t* cur = bufA; half_t* nxt = bufB;
    for (int l = 0; l < 3; ++l) {
      run_gemm(P[i_sa3 + 4 * l], P[i_sa3 + 4 * l + 1], wbuf, cur, nxt, couts[l], cin, M3, stream);
      run_bn_relu(nxt, P[i_sa3 + 4 * l + 2], P[i_sa3 + 4 * l + 3], mean, var, psum, psqs,
                  couts[l], M3, stream);
      half_t* t = cur; cur = nxt; nxt = t; cin = couts[l];
    }
    // final activations (pred3 input) now in bufB
  }
  // ---------------- pred3 ----------------
  run_gemm(P[i_pm3 + 0], P[i_pm3 + 1], wbuf, bufB, bufA, 512, 1024, M3, stream);
  run_bn_relu(bufA, P[i_pm3 + 2], P[i_pm3 + 3], mean, var, psum, psqs, 512, M3, stream);
  run_gemm(P[i_pm3 + 4], P[i_pm3 + 5], wbuf, bufA, bufB, 512, 512, M3, stream);
  colnorm_kernel<<<(M3 + 255) / 256, 256, 0, stream>>>(bufB, inv, 512, M3);
  rowmax_kernel<<<(B * 512 + 255) / 256, 256, 0, stream>>>(bufB, inv, out, B, 512, S2, 1024, 1536);
}
